// AEFIN_54357106098496
// MI455X (gfx1250) — compile-verified, matrix operand order checked by reading, not run on hardware
//
#include <hip/hip_runtime.h>
#include <math.h>

typedef _Float16 half_t;
typedef __attribute__((ext_vector_type(16))) _Float16 v16h;
typedef __attribute__((ext_vector_type(8)))  _Float16 v8h;
typedef __attribute__((ext_vector_type(8)))  float    v8f;
typedef __attribute__((ext_vector_type(4)))  int      v4i;

#define BSZ   64
#define SEQ   512
#define EIN   64
#define NFREQ 257
#define HDIM  1024   /* SEQ + PRED */
#define FC1N  3072

#define ACT_NONE 0
#define ACT_RELU 1
#define ACT_GELU 2
#define ACT_COS  3
#define ACT_SIN  4

#define OUT_F16  0
#define OUT_PRED 2

#ifndef __has_builtin
#define __has_builtin(x) 0
#endif

// CDNA5 async global->LDS copies (ASYNCcnt-tracked). Guarded: fall back to
// register staging if the builtin is absent so the TU always compiles.
#if __has_builtin(__builtin_amdgcn_global_load_async_to_lds_b128)
#define HAVE_ASYNC 1
#else
#define HAVE_ASYNC 0
#endif

#define ASG __attribute__((address_space(1)))
#define ASL __attribute__((address_space(3)))

__device__ __forceinline__ void cp_b128(const half_t* g, half_t* l) {
#if HAVE_ASYNC
  __builtin_amdgcn_global_load_async_to_lds_b128(
      (ASG v4i*)(void*)const_cast<half_t*>(g), (ASL v4i*)(void*)l, 0, 0);
#else
  *(v8h*)l = *(const v8h*)g;
#endif
}

__device__ __forceinline__ void cp_wait() {
#if HAVE_ASYNC
  asm volatile("s_wait_asynccnt 0x0" ::: "memory");
#endif
}

// ---------------------------------------------------------------------------
// Fragment loader: 16-bit A/B operand layout per CDNA5 ISA 7.12.2.
// lanes 0-15: halves 0..7 = K[kb..+7], halves 8..15 = K[kb+16..+23], kb=0
// lanes 16-31: same with kb=8   (kb = (lane>>4)*8, + ks*32 per K step)
// ---------------------------------------------------------------------------
__device__ __forceinline__ v16h frag_ld(const half_t* rowptr, int kb) {
  v8h lo = *(const v8h*)(rowptr + kb);
  v8h hi = *(const v8h*)(rowptr + kb + 16);
  v16h r;
#pragma unroll
  for (int i = 0; i < 8; ++i) { r[i] = lo[i]; r[i + 8] = hi[i]; }
  return r;
}

__device__ __forceinline__ v8f wmma16x16x32(v16h a, v16h b, v8f c) {
  return __builtin_amdgcn_wmma_f32_16x16x32_f16(false, a, false, b, (short)0, c,
                                                false, false);
}

// ---------------------------------------------------------------------------
// f32 -> f16 conversion
// ---------------------------------------------------------------------------
__global__ void cvt_f16_kernel(const float* __restrict__ s,
                               half_t* __restrict__ d, int n) {
  int i = blockIdx.x * 256 + threadIdx.x;
  if (i < n) d[i] = (half_t)s[i];
}

// ---------------------------------------------------------------------------
// DFT top-k filter. Block = (batch b, 8 channels); wave w handles channel w.
// Produces x_filt / norm (f16, [b,L,c]), mf transposed (f16, [(b*64+c), t]),
// and fills the xt half (cols 512..1023) of the MLP input buffer h.
// ---------------------------------------------------------------------------
__global__ __launch_bounds__(256) void dft_filter_kernel(
    const float* __restrict__ x, const int* __restrict__ topkp,
    half_t* __restrict__ xf16, half_t* __restrict__ nf16,
    half_t* __restrict__ mft, half_t* __restrict__ hbuf) {
  __shared__ float xs[8 * 512];
  __shared__ float ctab[512];
  __shared__ float stab[512];
  __shared__ float re[8 * 264];
  __shared__ float im[8 * 264];
  __shared__ float mg[8 * 264];
  __shared__ int fsel[8 * 32];

  const int tid = threadIdx.x;
  const int b = blockIdx.x >> 3;
  const int c0 = (blockIdx.x & 7) * 8;

  for (int i = tid; i < 512; i += 256) {
    float ang = 0.01227184630308513f * (float)i;  // 2*pi/512
    ctab[i] = cosf(ang);
    stab[i] = sinf(ang);
  }
  for (int i = tid; i < 4096; i += 256) {
    int c = i & 7, t = i >> 3;
    float v = x[((size_t)(b * 512 + t)) * 64 + c0 + c];
    xs[c * 512 + t] = v;
    hbuf[((size_t)(b * 64 + c0 + c)) * HDIM + 512 + t] = (half_t)v;
  }
  __syncthreads();

  const int wid = tid >> 5, lane = tid & 31;
  const float* xc = xs + wid * 512;

  // forward DFT (rfft): each lane owns freqs lane, lane+32, ...
  for (int f = lane; f < NFREQ; f += 32) {
    float sr = 0.f, si = 0.f;
    for (int t = 0; t < 512; ++t) {
      int idx = (f * t) & 511;
      float v = xc[t];
      sr += v * ctab[idx];
      si -= v * stab[idx];
    }
    re[wid * 264 + f] = sr;
    im[wid * 264 + f] = si;
    mg[wid * 264 + f] = sr * sr + si * si;  // monotone in |X|
  }
  __syncthreads();

  int KF = topkp ? topkp[0] : 20;
  if (KF > 32) KF = 32;
  if (KF < 1) KF = 1;

  // iterative top-k by wave argmax reduction
  for (int it = 0; it < KF; ++it) {
    float bv = -1.f;
    int bi = 0;
    for (int f = lane; f < NFREQ; f += 32) {
      float v = mg[wid * 264 + f];
      if (v > bv) { bv = v; bi = f; }
    }
    for (int off = 16; off >= 1; off >>= 1) {
      float ov = __shfl_xor(bv, off);
      int oi = __shfl_xor(bi, off);
      if (ov > bv || (ov == bv && oi < bi)) { bv = ov; bi = oi; }
    }
    if (lane == 0) {
      fsel[wid * 32 + it] = bi;
      mg[wid * 264 + bi] = -2.f;
    }
    __syncthreads();
  }

  // sparse inverse DFT (only KF freqs contribute)
  for (int t = lane; t < 512; t += 32) {
    float acc = 0.f;
    for (int j = 0; j < KF; ++j) {
      int f = fsel[wid * 32 + j];
      float w = (f == 0 || f == 256) ? 1.f : 2.f;
      int idx = (f * t) & 511;
      acc += w * (re[wid * 264 + f] * ctab[idx] - im[wid * 264 + f] * stab[idx]);
    }
    float xfv = acc * (1.f / 512.f);
    float xv = xc[t];
    size_t gidx = ((size_t)(b * 512 + t)) * 64 + c0 + wid;
    xf16[gidx] = (half_t)xfv;
    nf16[gidx] = (half_t)(xv - xfv);
    mft[((size_t)(b * 64 + c0 + wid)) * 512 + t] = (half_t)xfv;
  }
}

// ---------------------------------------------------------------------------
// Generic WMMA GEMM: C[M,N] = act(A[M,K] * B[N,K]^T + bias).
// Block tile 128(M) x 64(N), 8 waves 4x2, 32x32 per wave (2x2 WMMA).
// Double-buffered LDS tiles with async global->LDS staging.
// grid = (N/64, M/128). M%128==0, N%64==0, K%32==0 guaranteed by caller.
// ---------------------------------------------------------------------------
__global__ __launch_bounds__(256) void gemm_f16_kernel(
    const half_t* __restrict__ A, int lda, const half_t* __restrict__ Bw,
    int ldb, const float* __restrict__ bias, const float* __restrict__ gatep,
    int K, int act, int outmode, void* __restrict__ outp, int ldc, int coloff) {
  __shared__ half_t At[2][128 * 40];
  __shared__ half_t Bt[2][64 * 40];

  const int tid = threadIdx.x;
  const int wid = tid >> 5, lane = tid & 31;
  const int wm = wid & 3, wn = wid >> 2;
  const int m0 = blockIdx.y * 128, n0 = blockIdx.x * 64;
  const int lhalf = lane >> 4, lcol = lane & 15;

  // straight-line per-thread tile-load coordinates (b128 = 8-half chunks)
  const int ar0 = tid >> 2;             // A chunk 0 row (0..63)
  const int ar1 = (tid + 256) >> 2;     // A chunk 1 row (64..127)
  const int ac = (tid & 3) * 8;         // A/B chunk column (halves)
  const half_t* arow0 = A + (size_t)(m0 + ar0) * lda + ac;
  const half_t* arow1 = A + (size_t)(m0 + ar1) * lda + ac;
  const half_t* brow = Bw + (size_t)(n0 + (tid >> 2)) * ldb + ac;
  const int aoff0 = ar0 * 40 + ac, aoff1 = ar1 * 40 + ac;
  const int boff = (tid >> 2) * 40 + ac;

  v8f acc[2][2] = {};
  const int nk = K >> 5;

  // prologue: stage tile 0
  cp_b128(arow0, At[0] + aoff0);
  cp_b128(arow1, At[0] + aoff1);
  cp_b128(brow, Bt[0] + boff);

  for (int t = 0; t < nk; ++t) {
    cp_wait();
    __syncthreads();  // tile t resident in buffer t&1

    if (t + 1 < nk) {  // stage tile t+1 into the other buffer (overlaps WMMA)
      const int k1 = (t + 1) * 32;
      half_t* at = At[(t + 1) & 1];
      half_t* bt = Bt[(t + 1) & 1];
      cp_b128(arow0 + k1, at + aoff0);
      cp_b128(arow1 + k1, at + aoff1);
      cp_b128(brow + k1, bt + boff);
    }

    const half_t* at = At[t & 1];
    const half_t* bt = Bt[t & 1];
    const int kb = lhalf * 8;
    v16h a0 = frag_ld(at + (wm * 32 + lcol) * 40, kb);
    v16h a1 = frag_ld(at + (wm * 32 + 16 + lcol) * 40, kb);
    v16h b0 = frag_ld(bt + (wn * 32 + lcol) * 40, kb);
    v16h b1 = frag_ld(bt + (wn * 32 + 16 + lcol) * 40, kb);
    acc[0][0] = wmma16x16x32(a0, b0, acc[0][0]);
    acc[0][1] = wmma16x16x32(a0, b1, acc[0][1]);
    acc[1][0] = wmma16x16x32(a1, b0, acc[1][0]);
    acc[1][1] = wmma16x16x32(a1, b1, acc[1][1]);
  }

  float gt = 0.f;
  if (gatep) gt = 1.f / (1.f + expf(-gatep[0]));

#pragma unroll
  for (int im = 0; im < 2; ++im) {
#pragma unroll
    for (int jn = 0; jn < 2; ++jn) {
      int nglob = n0 + wn * 32 + jn * 16 + lcol;
      float bsv = bias ? bias[nglob] : 0.f;
#pragma unroll
      for (int r = 0; r < 8; ++r) {
        int m = m0 + wm * 32 + im * 16 + r + 8 * lhalf;  // C/D layout 7.12.2
        float v = acc[im][jn][r] + bsv;
        if (act == ACT_RELU) {
          v = fmaxf(v, 0.f);
        } else if (act == ACT_GELU) {
          v = 0.5f * v * (1.f + erff(v * 0.70710678f)) * (1.f - gt);
        } else if (act == ACT_COS) {
          v = gt * cosf(v);
        } else if (act == ACT_SIN) {
          v = gt * sinf(v);
        }
        if (outmode == OUT_F16) {
          ((half_t*)outp)[(size_t)m * ldc + coloff + nglob] = (half_t)v;
        } else {  // OUT_PRED: pred_main[b, t, c] from row m=(b*64+c), col n=t
          ((float*)outp)[((size_t)(m >> 6) * 512 + nglob) * 64 + (m & 63)] = v;
        }
      }
    }
  }
}

// ---------------------------------------------------------------------------
// Fused single-head attention (flash-style online softmax).
// grid = (4, B); block = 256 = 8 waves; wave owns 16 query rows.
// Q = x_filt proj, K/V = norm proj (f16 [B*L, 64]); out = f32 [B, L, 64].
// ---------------------------------------------------------------------------
__global__ __launch_bounds__(256) void attn_kernel(
    const half_t* __restrict__ qg, const half_t* __restrict__ kg,
    const half_t* __restrict__ vg, const half_t* __restrict__ outw,
    const float* __restrict__ outb, float* __restrict__ outp) {
  __shared__ half_t kt[64 * 72];       // K tile, [key][E]
  __shared__ half_t vt[64 * 72];       // V tile transposed, [E][key]
  __shared__ half_t pb[8 * 16 * 72];   // per-wave P / O staging

  const int b = blockIdx.y;
  const int tid = threadIdx.x;
  const int wid = tid >> 5, lane = tid & 31;
  const int lhalf = lane >> 4, lcol = lane & 15;
  const int qr0 = blockIdx.x * 128 + wid * 16;
  half_t* pw = pb + wid * 16 * 72;

  // Q fragments (16 rows x 64 E) held in registers for the whole kernel
  const half_t* qrow = qg + ((size_t)(b * 512 + qr0 + lcol)) * 64;
  v16h qa[2];
#pragma unroll
  for (int ks = 0; ks < 2; ++ks) qa[ks] = frag_ld(qrow, lhalf * 8 + ks * 32);

  v8f o[4] = {};
  float mrun[8], lrun[8];
#pragma unroll
  for (int r = 0; r < 8; ++r) { mrun[r] = -1e30f; lrun[r] = 0.f; }

  // per-thread K-tile chunk coords (64 rows x 8 b128-chunks = 2 per thread)
  const int kr0 = tid >> 3, kr1 = (tid + 256) >> 3;
  const int kc = (tid & 7) * 8;

  for (int kb0 = 0; kb0 < 512; kb0 += 64) {
    __syncthreads();
    const half_t* kbase = kg + ((size_t)(b * 512 + kb0)) * 64;
    cp_b128(kbase + (size_t)kr0 * 64 + kc, kt + kr0 * 72 + kc);
    cp_b128(kbase + (size_t)kr1 * 64 + kc, kt + kr1 * 72 + kc);
    for (int i = tid; i < 4096; i += 256) {  // V tile transposed (scalar)
      int key = i >> 6, e = i & 63;
      vt[e * 72 + key] = vg[((size_t)(b * 512 + kb0 + key)) * 64 + e];
    }
    cp_wait();
    __syncthreads();

    // S = Q * K^T (scaled)
    v8f s[4];
#pragma unroll
    for (int j = 0; j < 4; ++j) {
      v8f z = {};
#pragma unroll
      for (int ks = 0; ks < 2; ++ks) {
        v16h bf = frag_ld(kt + (j * 16 + lcol) * 72, lhalf * 8 + ks * 32);
        z = wmma16x16x32(qa[ks], bf, z);
      }
      s[j] = z;
    }

    // online softmax, row stats per C-frag row (m = r + 8*lhalf)
#pragma unroll
    for (int r = 0; r < 8; ++r) {
      float rowmax = -1e30f;
#pragma unroll
      for (int j = 0; j < 4; ++j) {
        s[j][r] *= 0.125f;  // 1/sqrt(64)
        rowmax = fmaxf(rowmax, s[j][r]);
      }
      for (int off = 8; off >= 1; off >>= 1)
        rowmax = fmaxf(rowmax, __shfl_xor(rowmax, off));
      float newm = fmaxf(mrun[r], rowmax);
      float fs = expf(mrun[r] - newm);
      lrun[r] *= fs;
#pragma unroll
      for (int j = 0; j < 4; ++j) o[j][r] *= fs;
      float rs = 0.f;
#pragma unroll
      for (int j = 0; j < 4; ++j) {
        float p = expf(s[j][r] - newm);
        s[j][r] = p;
        rs += p;
      }
      for (int off = 8; off >= 1; off >>= 1) rs += __shfl_xor(rs, off);
      lrun[r] += rs;
      mrun[r] = newm;
      int ml = r + 8 * lhalf;
#pragma unroll
      for (int j = 0; j < 4; ++j)
        pw[ml * 72 + j * 16 + lcol] = (half_t)s[j][r];
    }

    // O += P * V
#pragma unroll
    for (int ks = 0; ks < 2; ++ks) {
      v16h pa = frag_ld(pw + lcol * 72, lhalf * 8 + ks * 32);
#pragma unroll
      for (int j = 0; j < 4; ++j) {
        v16h vf = frag_ld(vt + (j * 16 + lcol) * 72, lhalf * 8 + ks * 32);
        o[j] = wmma16x16x32(pa, vf, o[j]);
      }
    }
  }

  // normalize and stage O for the output projection
#pragma unroll
  for (int r = 0; r < 8; ++r) {
    float inv = 1.f / lrun[r];
    int ml = r + 8 * lhalf;
#pragma unroll
    for (int j = 0; j < 4; ++j) {
      o[j][r] *= inv;
      pw[ml * 72 + j * 16 + lcol] = (half_t)o[j][r];
    }
  }

  // out = O * out_w^T + out_b
  v8f res[4] = {};
#pragma unroll
  for (int ks = 0; ks < 2; ++ks) {
    v16h oa = frag_ld(pw + lcol * 72, lhalf * 8 + ks * 32);
#pragma unroll
    for (int j = 0; j < 4; ++j) {
      v16h wf = frag_ld(outw + (size_t)(j * 16 + lcol) * 64, lhalf * 8 + ks * 32);
      res[j] = wmma16x16x32(oa, wf, res[j]);
    }
  }
#pragma unroll
  for (int j = 0; j < 4; ++j) {
    int n = j * 16 + lcol;
    float bsv = outb[n];
#pragma unroll
    for (int r = 0; r < 8; ++r) {
      int m = qr0 + r + 8 * lhalf;
      outp[((size_t)(b * 512 + m)) * 64 + n] = res[j][r] + bsv;
    }
  }
}

// ---------------------------------------------------------------------------
extern "C" void kernel_launch(void* const* d_in, const int* in_sizes, int n_in,
                              void* d_out, int out_size, void* d_ws,
                              size_t ws_size, hipStream_t stream) {
  (void)in_sizes; (void)n_in; (void)out_size; (void)ws_size;

  const float* batch_x  = (const float*)d_in[0];
  const float* in_projw = (const float*)d_in[1];
  const float* in_projb = (const float*)d_in[2];
  const float* out_w    = (const float*)d_in[3];
  const float* out_b    = (const float*)d_in[4];
  const float* Wp       = (const float*)d_in[5];
  const float* bp       = (const float*)d_in[6];
  const float* Wg       = (const float*)d_in[7];
  const float* bg       = (const float*)d_in[8];
  const float* gate     = (const float*)d_in[9];
  const float* fc1_w    = (const float*)d_in[10];
  const float* fc1_b    = (const float*)d_in[11];
  const float* fc2_w    = (const float*)d_in[12];
  const float* fc2_b    = (const float*)d_in[13];
  const int*   topk     = (const int*)d_in[14];
  float* out = (float*)d_out;

  char* ws = (char*)d_ws;
  size_t off = 0;
  auto walloc = [&](size_t elems) -> half_t* {
    half_t* p = (half_t*)(ws + off);
    off = (off + elems * sizeof(half_t) + 255) & ~(size_t)255;
    return p;
  };

  half_t* projw_h = walloc(192 * 64);
  half_t* outw_h  = walloc(64 * 64);
  half_t* wp_h    = walloc(128 * 512);
  half_t* wg_h    = walloc(256 * 512);
  half_t* fc1_h   = walloc((size_t)FC1N * HDIM);
  half_t* fc2_h   = walloc((size_t)512 * FC1N);
  half_t* xf16    = walloc((size_t)BSZ * SEQ * EIN);
  half_t* nf16    = walloc((size_t)BSZ * SEQ * EIN);
  half_t* mft     = walloc((size_t)BSZ * EIN * SEQ);
  half_t* hbuf    = walloc((size_t)4096 * HDIM);
  half_t* hid     = walloc((size_t)4096 * FC1N);
  half_t* qh      = walloc((size_t)BSZ * SEQ * EIN);
  half_t* kh      = walloc((size_t)BSZ * SEQ * EIN);
  half_t* vh      = walloc((size_t)BSZ * SEQ * EIN);

  auto cvt = [&](const float* s, half_t* d, int n) {
    cvt_f16_kernel<<<(n + 255) / 256, 256, 0, stream>>>(s, d, n);
  };
  cvt(in_projw, projw_h, 192 * 64);
  cvt(out_w, outw_h, 64 * 64);
  cvt(Wp, wp_h, 128 * 512);
  cvt(Wg, wg_h, 256 * 512);
  cvt(fc1_w, fc1_h, FC1N * HDIM);
  cvt(fc2_w, fc2_h, 512 * FC1N);

  // 1) DFT top-k filter: x_filt / norm / mf^T / xt half of h
  dft_filter_kernel<<<BSZ * 8, 256, 0, stream>>>(batch_x, topk, xf16, nf16,
                                                 mft, hbuf);

  // 2) QKV projections: [32768,64] x [64,64]^T
  gemm_f16_kernel<<<dim3(1, 256), 256, 0, stream>>>(
      xf16, 64, projw_h, 64, in_projb, nullptr, 64, ACT_NONE, OUT_F16, qh, 64, 0);
  gemm_f16_kernel<<<dim3(1, 256), 256, 0, stream>>>(
      nf16, 64, projw_h + 64 * 64, 64, in_projb + 64, nullptr, 64, ACT_NONE,
      OUT_F16, kh, 64, 0);
  gemm_f16_kernel<<<dim3(1, 256), 256, 0, stream>>>(
      nf16, 64, projw_h + 128 * 64, 64, in_projb + 128, nullptr, 64, ACT_NONE,
      OUT_F16, vh, 64, 0);

  // 3) FAN layer -> h cols [0,128)=gt*cos, [128,256)=gt*sin, [256,512)=(1-gt)*gelu
  gemm_f16_kernel<<<dim3(2, 32), 256, 0, stream>>>(
      mft, 512, wp_h, 512, bp, gate, 512, ACT_COS, OUT_F16, hbuf, HDIM, 0);
  gemm_f16_kernel<<<dim3(2, 32), 256, 0, stream>>>(
      mft, 512, wp_h, 512, bp, gate, 512, ACT_SIN, OUT_F16, hbuf, HDIM, 128);
  gemm_f16_kernel<<<dim3(4, 32), 256, 0, stream>>>(
      mft, 512, wg_h, 512, bg, gate, 512, ACT_GELU, OUT_F16, hbuf, HDIM, 256);

  // 4) MLP: fc1 (relu) then fc2 (scatter-transposed into pred_main)
  gemm_f16_kernel<<<dim3(FC1N / 64, 32), 256, 0, stream>>>(
      hbuf, HDIM, fc1_h, HDIM, fc1_b, nullptr, HDIM, ACT_RELU, OUT_F16, hid,
      FC1N, 0);
  gemm_f16_kernel<<<dim3(512 / 64, 32), 256, 0, stream>>>(
      hid, FC1N, fc2_h, FC1N, fc2_b, nullptr, FC1N, ACT_NONE, OUT_PRED,
      out + (size_t)BSZ * SEQ * EIN, 0, 0);

  // 5) Fused attention -> d_out[0 .. B*L*E)
  attn_kernel<<<dim3(4, BSZ), 256, 0, stream>>>(qh, kh, vh, outw_h, out_b, out);
}
